// ResidualGraphBlock_12790412607605
// MI455X (gfx1250) — matrix operand order repaired and checked by reading, not compiled, not run on hardware
//
#include <hip/hip_runtime.h>
#include <hip/hip_bf16.h>

// ---------------------------------------------------------------------------
// Types for CDNA5 WMMA (wave32): v_wmma_f32_16x16x32_bf16
// ---------------------------------------------------------------------------
typedef __attribute__((ext_vector_type(16))) __bf16 v16bf;
typedef __attribute__((ext_vector_type(8)))  float  v8f;
typedef __attribute__((ext_vector_type(4)))  unsigned int u32x4;
typedef __attribute__((ext_vector_type(2)))  unsigned int u32x2;

#define IN_F 128
#define OUT_F 128
#define HEADS 8
#define DDIM 16
#define GATE_F 64

// float -> bf16 bits (round-to-nearest-even), pure bit ops
__device__ __host__ inline unsigned short f2bf_bits(float f) {
    union { float f; unsigned u; } v; v.f = f;
    unsigned u = v.u + 0x7fffu + ((v.u >> 16) & 1u);
    return (unsigned short)(u >> 16);
}
__device__ __host__ inline __bf16 f2bf(float f) {
    union { unsigned short s; __bf16 b; } r; r.s = f2bf_bits(f);
    return r.b;
}

// Monotone order-preserving key for float atomicMax over full range
__device__ inline unsigned fkey(float f) {
    unsigned u = __float_as_uint(f);
    return (u & 0x80000000u) ? ~u : (u | 0x80000000u);
}
__device__ inline float fkey_dec(unsigned u) {
    unsigned b = (u & 0x80000000u) ? (u & 0x7fffffffu) : ~u;
    return __uint_as_float(b);
}

__device__ inline float eluf(float x) { return x > 0.f ? x : (__expf(x) - 1.f); }

// ---------------------------------------------------------------------------
// WMMA fragment helpers (bf16 16x16x32, layouts per CDNA5 ISA)
// A 16x32: lanes 0-15 -> M=0..15, elems 0..7 => K=kb+0..7, 8..15 => K=kb+16..23
//          lanes 16-31 same M, +8 on K within each half.
// B 32x16: lane&15 = N column; K order as A (lanes 0-15: K=kb+0..7,16..23; lanes 16-31: K=kb+8..15,24..31)
// C/D    : VGPR j -> M = j + (lane>=16 ? 8 : 0), N = lane&15
// ---------------------------------------------------------------------------
union frag_u { v16bf v; u32x4 q[2]; };

// Vectorized A-fragment load: two 16-byte loads (global_load_b128 / ds_load_b128)
__device__ inline v16bf load_a_frag(const __bf16* X, int ld, int row0, int kb, int lane) {
    int m   = lane & 15;
    int hi8 = (lane >> 4) << 3;       // 0 or 8
    const __bf16* rp = X + (size_t)(row0 + m) * ld + kb + hi8;
    frag_u u;
    u.q[0] = *(const u32x4*)(rp);
    u.q[1] = *(const u32x4*)(rp + 16);
    return u.v;
}

// B fragments come pre-packed: P[((tile*nkb + kb32)*32 + lane)*16 + e]
__device__ inline v16bf load_b_packed(const __bf16* P, int nkb, int tile, int kb32, int lane) {
    const __bf16* rp = P + (((size_t)(tile * nkb + kb32) * 32 + lane) << 4);
    frag_u u;
    u.q[0] = *(const u32x4*)(rp);
    u.q[1] = *(const u32x4*)(rp + 8);
    return u.v;
}

__device__ inline v8f wmma_bf16(v16bf a, v16bf b, v8f c) {
    return __builtin_amdgcn_wmma_f32_16x16x32_bf16(false, a, false, b, (short)0, c,
                                                   false, false);
}

// ---------------------------------------------------------------------------
// Pack a row-major f32 weight [K x ncols] into bf16 B-fragment order:
// out[((t*nkb + kb)*32 + lane)*16 + e] = W[(kb*32 + ((lane>>4)<<3) + (e&7) + ((e>>3)<<4))*ncols
//                                          + t*16 + (lane&15)]
// ---------------------------------------------------------------------------
__global__ void pack_b_kernel(const float* __restrict__ W, __bf16* __restrict__ P,
                              int K, int ncols) {
    int i = blockIdx.x * 256 + threadIdx.x;
    int total = K * ncols;
    if (i >= total) return;
    int nkb = K >> 5;
    int e    = i & 15;
    int lane = (i >> 4) & 31;
    int kb   = (i >> 9) % nkb;
    int t    = i / (nkb << 9);
    int k = (kb << 5) + ((lane >> 4) << 3) + (e & 7) + ((e >> 3) << 4);
    int n = (t << 4) + (lane & 15);
    P[i] = f2bf(W[(size_t)k * ncols + n]);
}

// ---------------------------------------------------------------------------
// K0: LayerNorm1, wave-per-node, lane handles 4 features (128 = 32*4)
// ---------------------------------------------------------------------------
__global__ void ln1_kernel(const float* __restrict__ h, const float* __restrict__ g,
                           const float* __restrict__ b, float* __restrict__ hn,
                           __bf16* __restrict__ hnb, int N) {
    int wave = threadIdx.x >> 5, lane = threadIdx.x & 31;
    int node = blockIdx.x * 8 + wave;
    if (node >= N) return;
    const float4 x = ((const float4*)(h + (size_t)node * IN_F))[lane];
    float s  = x.x + x.y + x.z + x.w;
    float s2 = x.x * x.x + x.y * x.y + x.z * x.z + x.w * x.w;
#pragma unroll
    for (int m = 1; m < 32; m <<= 1) { s += __shfl_xor(s, m, 32); s2 += __shfl_xor(s2, m, 32); }
    float mu  = s * (1.f / IN_F);
    float var = s2 * (1.f / IN_F) - mu * mu;
    float rs  = rsqrtf(var + 1e-5f);
    int f = lane * 4;
    float4 o;
    o.x = (x.x - mu) * rs * g[f + 0] + b[f + 0];
    o.y = (x.y - mu) * rs * g[f + 1] + b[f + 1];
    o.z = (x.z - mu) * rs * g[f + 2] + b[f + 2];
    o.w = (x.w - mu) * rs * g[f + 3] + b[f + 3];
    ((float4*)(hn + (size_t)node * IN_F))[lane] = o;
    u32x2 pk;
    pk.x = (unsigned)f2bf_bits(o.x) | ((unsigned)f2bf_bits(o.y) << 16);
    pk.y = (unsigned)f2bf_bits(o.z) | ((unsigned)f2bf_bits(o.w) << 16);
    *(u32x2*)(hnb + (size_t)node * IN_F + f) = pk;
}

// ---------------------------------------------------------------------------
// K1: fused front GEMMs: ft = hn@W_attn ; p = relu(hn@W_pool + b_pool) ;
//     gself = hn@W_gself.  Block = 8 waves on one 16-row stripe; each wave
//     loads its 4 A fragments ONCE and loops over its 2-3 of 20 col tiles.
//     B fragments are hoisted ahead of the WMMA chain for latency overlap.
// ---------------------------------------------------------------------------
__global__ void gemm_front_kernel(const __bf16* __restrict__ hnb,
                                  const __bf16* __restrict__ WaP,
                                  const __bf16* __restrict__ WpP,
                                  const __bf16* __restrict__ WgP,
                                  const float* __restrict__ b_pool,
                                  float* __restrict__ ft, float* __restrict__ pOut,
                                  float* __restrict__ gs) {
    int lane = threadIdx.x & 31;
    int wave = threadIdx.x >> 5;
    int r0 = blockIdx.x * 16;

    // warm L2 for the packed weights (global_prefetch_b8)
    __builtin_prefetch(WaP + (lane << 6), 0, 1);
    __builtin_prefetch(WpP + (lane << 6), 0, 1);

    v16bf a0 = load_a_frag(hnb, IN_F, r0, 0,  lane);
    v16bf a1 = load_a_frag(hnb, IN_F, r0, 32, lane);
    v16bf a2 = load_a_frag(hnb, IN_F, r0, 64, lane);
    v16bf a3 = load_a_frag(hnb, IN_F, r0, 96, lane);

    int col16 = lane & 15;
    int mof   = (lane >> 4) << 3;

    for (int tt = wave; tt < 20; tt += 8) {
        const __bf16* P; int tile;
        if (tt < 8)       { P = WaP; tile = tt; }
        else if (tt < 16) { P = WpP; tile = tt - 8; }
        else              { P = WgP; tile = tt - 16; }
        // hoist all B loads ahead of the WMMA chain
        v16bf b0 = load_b_packed(P, 4, tile, 0, lane);
        v16bf b1 = load_b_packed(P, 4, tile, 1, lane);
        v16bf b2 = load_b_packed(P, 4, tile, 2, lane);
        v16bf b3 = load_b_packed(P, 4, tile, 3, lane);
        v8f acc = {};
        acc = wmma_bf16(a0, b0, acc);
        acc = wmma_bf16(a1, b1, acc);
        acc = wmma_bf16(a2, b2, acc);
        acc = wmma_bf16(a3, b3, acc);
        int col = tile * 16 + col16;
        if (tt < 8) {
#pragma unroll
            for (int j = 0; j < 8; ++j) ft[(size_t)(r0 + mof + j) * OUT_F + col] = acc[j];
        } else if (tt < 16) {
#pragma unroll
            for (int j = 0; j < 8; ++j)
                pOut[(size_t)(r0 + mof + j) * IN_F + col] = fmaxf(acc[j] + b_pool[col], 0.f);
        } else {
#pragma unroll
            for (int j = 0; j < 8; ++j) gs[(size_t)(r0 + mof + j) * GATE_F + col] = acc[j];
        }
    }
}

// ---------------------------------------------------------------------------
// K2: edge pass 1 — attention scores + segment max (keys) + pm segment max.
// ---------------------------------------------------------------------------
__global__ void edge_pass1_kernel(const int* __restrict__ src, const int* __restrict__ dst,
                                  const float* __restrict__ ft, const float* __restrict__ pOut,
                                  float* __restrict__ esc, unsigned* __restrict__ mkey,
                                  int* __restrict__ pmbits, int E) {
    int wave = threadIdx.x >> 5, lane = threadIdx.x & 31;
    long e = (long)blockIdx.x * 8 + wave;
    if (e >= E) return;
    int s = src[e], d = dst[e];
    int head = lane >> 2, q = lane & 3;
    int off = head * DDIM + q * 4;
    float4 a = *(const float4*)(ft + (size_t)s * OUT_F + off);
    float4 b = *(const float4*)(ft + (size_t)d * OUT_F + off);
    float pr = a.x * b.x + a.y * b.y + a.z * b.z + a.w * b.w;
    pr += __shfl_xor(pr, 1, 32);
    pr += __shfl_xor(pr, 2, 32);
    if (q == 0) {
        float sc = pr * 0.25f;                 // 1/sqrt(D), D=16
        esc[(size_t)e * HEADS + head] = sc;
        atomicMax(&mkey[(size_t)d * HEADS + head], fkey(sc));
    }
    // pm = segment_max over in-neighbors of relu'd p (>=0 -> raw int-bit max OK)
    const float* ps = pOut + (size_t)s * IN_F;
#pragma unroll
    for (int r = 0; r < 4; ++r) {
        int f = lane + 32 * r;
        atomicMax(&pmbits[(size_t)d * IN_F + f], __float_as_int(ps[f]));
    }
}

// ---------------------------------------------------------------------------
// K3: edge pass 2 — ex = exp(e - m[dst]); z[dst] += ex. Thread per (edge,head)
// ---------------------------------------------------------------------------
__global__ void edge_pass2_kernel(const int* __restrict__ dst, float* __restrict__ esc,
                                  const unsigned* __restrict__ mkey, float* __restrict__ z,
                                  long EH) {
    long i = (long)blockIdx.x * 256 + threadIdx.x;
    if (i >= EH) return;
    long e = i >> 3; int hh = (int)(i & 7);
    int d = dst[e];
    float m  = fkey_dec(mkey[(size_t)d * HEADS + hh]);
    float ex = __expf(esc[i] - m);
    esc[i] = ex;
    atomicAdd(&z[(size_t)d * HEADS + hh], ex);
}

// ---------------------------------------------------------------------------
// K4: edge pass 3 — agg[dst] += ft[src] * alpha. Wave per edge.
// ---------------------------------------------------------------------------
__global__ void edge_pass3_kernel(const int* __restrict__ src, const int* __restrict__ dst,
                                  const float* __restrict__ ft, const float* __restrict__ esc,
                                  const float* __restrict__ z, float* __restrict__ agg, int E) {
    int wave = threadIdx.x >> 5, lane = threadIdx.x & 31;
    long e = (long)blockIdx.x * 8 + wave;
    if (e >= E) return;
    int s = src[e], d = dst[e];
    int head = lane >> 2, q = lane & 3;
    int off = head * DDIM + q * 4;
    float alpha = esc[(size_t)e * HEADS + head] / z[(size_t)d * HEADS + head];
    const float* fs = ft + (size_t)s * OUT_F + off;
    float* ag = agg + (size_t)d * OUT_F + off;
#pragma unroll
    for (int i = 0; i < 4; ++i) atomicAdd(&ag[i], fs[i] * alpha);
}

// ---------------------------------------------------------------------------
// K5: node epilogue — gate WMMA, gate reduce, elu*gate, W_hr WMMA + residual,
//     LN2, W_si WMMA + elu + residual. 16 nodes / block, 8 waves, LDS staged.
// ---------------------------------------------------------------------------
__global__ void node_epilogue_kernel(const float* __restrict__ hn, const float* __restrict__ agg,
                                     const int* __restrict__ pmbits, const float* __restrict__ gs,
                                     const __bf16* __restrict__ WgnP, const float* __restrict__ b_gate,
                                     const float* __restrict__ W_gr, const float* __restrict__ b_gr,
                                     const __bf16* __restrict__ WhrP, const float* __restrict__ b_hr,
                                     const float* __restrict__ ln2g, const float* __restrict__ ln2b,
                                     const __bf16* __restrict__ WsiP, const float* __restrict__ b_si,
                                     float* __restrict__ out) {
    __shared__ float  s_hn[16][IN_F];     // post-LN1 residual
    __shared__ __bf16 s_pmb[16][IN_F];    // pm in bf16
    __shared__ float  s_gf[16][GATE_F];   // gate features
    __shared__ float  s_gate[16][HEADS];
    __shared__ __bf16 s_hb[16][OUT_F];    // bf16 staging: h_att, then LN2(h2)
    __shared__ float  s_h2[16][OUT_F];    // f32: h2, then LN2(h2)
    __shared__ float  s_mu[16], s_rs[16];

    int t = threadIdx.x, lane = t & 31, wave = t >> 5;
    int r0 = blockIdx.x * 16;

    __builtin_prefetch(WhrP + (lane << 6), 0, 1);
    __builtin_prefetch(WsiP + (lane << 6), 0, 1);

    for (int idx = t; idx < 16 * IN_F; idx += 256) {
        int n = idx >> 7, f = idx & 127;
        s_hn[n][f]  = hn[(size_t)(r0 + n) * IN_F + f];
        s_pmb[n][f] = f2bf(__int_as_float(pmbits[(size_t)(r0 + n) * IN_F + f]));
    }
    __syncthreads();

    // gate_feat = pm @ W_gneigh + gself + b_gate   (waves 0..3, 4 col tiles)
    if (wave < 4) {
        v16bf b0 = load_b_packed(WgnP, 4, wave, 0, lane);
        v16bf b1 = load_b_packed(WgnP, 4, wave, 1, lane);
        v16bf b2 = load_b_packed(WgnP, 4, wave, 2, lane);
        v16bf b3 = load_b_packed(WgnP, 4, wave, 3, lane);
        v8f acc = {};
        acc = wmma_bf16(load_a_frag(&s_pmb[0][0], IN_F, 0, 0,  lane), b0, acc);
        acc = wmma_bf16(load_a_frag(&s_pmb[0][0], IN_F, 0, 32, lane), b1, acc);
        acc = wmma_bf16(load_a_frag(&s_pmb[0][0], IN_F, 0, 64, lane), b2, acc);
        acc = wmma_bf16(load_a_frag(&s_pmb[0][0], IN_F, 0, 96, lane), b3, acc);
        int col = wave * 16 + (lane & 15), mof = (lane >> 4) << 3;
#pragma unroll
        for (int j = 0; j < 8; ++j) {
            int row = mof + j;
            s_gf[row][col] = acc[j] + gs[(size_t)(r0 + row) * GATE_F + col] + b_gate[col];
        }
    }
    __syncthreads();

    // gate = sigmoid(gate_feat @ W_gr + b_gr)  (64x8, tiny -> VALU)
    if (t < 128) {
        int n = t >> 3, hh = t & 7;
        float a = 0.f;
#pragma unroll
        for (int k = 0; k < GATE_F; ++k) a += s_gf[n][k] * W_gr[k * HEADS + hh];
        s_gate[n][hh] = 1.f / (1.f + __expf(-(a + b_gr[hh])));
    }
    __syncthreads();

    // h_att = elu(agg) * gate -> bf16
    for (int idx = t; idx < 16 * OUT_F; idx += 256) {
        int n = idx >> 7, f = idx & 127;
        float av = agg[(size_t)(r0 + n) * OUT_F + f];
        s_hb[n][f] = f2bf(eluf(av) * s_gate[n][f >> 4]);
    }
    __syncthreads();

    // h2 = h_att @ W_hr + b_hr + hn   (8 waves, 8 col tiles)
    {
        v16bf b0 = load_b_packed(WhrP, 4, wave, 0, lane);
        v16bf b1 = load_b_packed(WhrP, 4, wave, 1, lane);
        v16bf b2 = load_b_packed(WhrP, 4, wave, 2, lane);
        v16bf b3 = load_b_packed(WhrP, 4, wave, 3, lane);
        v8f acc = {};
        acc = wmma_bf16(load_a_frag(&s_hb[0][0], OUT_F, 0, 0,  lane), b0, acc);
        acc = wmma_bf16(load_a_frag(&s_hb[0][0], OUT_F, 0, 32, lane), b1, acc);
        acc = wmma_bf16(load_a_frag(&s_hb[0][0], OUT_F, 0, 64, lane), b2, acc);
        acc = wmma_bf16(load_a_frag(&s_hb[0][0], OUT_F, 0, 96, lane), b3, acc);
        int col = wave * 16 + (lane & 15), mof = (lane >> 4) << 3;
#pragma unroll
        for (int j = 0; j < 8; ++j) {
            int row = mof + j;
            s_h2[row][col] = acc[j] + b_hr[col] + s_hn[row][col];
        }
    }
    __syncthreads();

    // LN2 statistics
    if (t < 16) {
        float s = 0.f, s2 = 0.f;
        for (int f = 0; f < OUT_F; ++f) { float v = s_h2[t][f]; s += v; s2 += v * v; }
        float mu = s * (1.f / OUT_F);
        float var = s2 * (1.f / OUT_F) - mu * mu;
        s_mu[t] = mu; s_rs[t] = rsqrtf(var + 1e-5f);
    }
    __syncthreads();

    for (int idx = t; idx < 16 * OUT_F; idx += 256) {
        int n = idx >> 7, f = idx & 127;
        float v = (s_h2[n][f] - s_mu[n]) * s_rs[n] * ln2g[f] + ln2b[f];
        s_h2[n][f] = v;           // LN2(h2), residual for final add
        s_hb[n][f] = f2bf(v);
    }
    __syncthreads();

    // out = elu(h2n @ W_si + b_si) + h2n
    {
        v16bf b0 = load_b_packed(WsiP, 4, wave, 0, lane);
        v16bf b1 = load_b_packed(WsiP, 4, wave, 1, lane);
        v16bf b2 = load_b_packed(WsiP, 4, wave, 2, lane);
        v16bf b3 = load_b_packed(WsiP, 4, wave, 3, lane);
        v8f acc = {};
        acc = wmma_bf16(load_a_frag(&s_hb[0][0], OUT_F, 0, 0,  lane), b0, acc);
        acc = wmma_bf16(load_a_frag(&s_hb[0][0], OUT_F, 0, 32, lane), b1, acc);
        acc = wmma_bf16(load_a_frag(&s_hb[0][0], OUT_F, 0, 64, lane), b2, acc);
        acc = wmma_bf16(load_a_frag(&s_hb[0][0], OUT_F, 0, 96, lane), b3, acc);
        int col = wave * 16 + (lane & 15), mof = (lane >> 4) << 3;
#pragma unroll
        for (int j = 0; j < 8; ++j) {
            int row = mof + j;
            float o = eluf(acc[j] + b_si[col]) + s_h2[row][col];
            out[(size_t)(r0 + row) * OUT_F + col] = o;
        }
    }
}

// ---------------------------------------------------------------------------
// Host launcher
// ---------------------------------------------------------------------------
extern "C" void kernel_launch(void* const* d_in, const int* in_sizes, int n_in,
                              void* d_out, int out_size, void* d_ws, size_t ws_size,
                              hipStream_t stream) {
    const float* h      = (const float*)d_in[0];
    const int*   src    = (const int*)d_in[1];
    const int*   dst    = (const int*)d_in[2];
    const float* ln1_g  = (const float*)d_in[3];
    const float* ln1_b  = (const float*)d_in[4];
    const float* W_attn = (const float*)d_in[5];
    const float* W_pool = (const float*)d_in[6];
    const float* b_pool = (const float*)d_in[7];
    const float* W_gself= (const float*)d_in[8];
    const float* W_gneigh=(const float*)d_in[9];
    const float* b_gate = (const float*)d_in[10];
    const float* W_gr   = (const float*)d_in[11];
    const float* b_gr   = (const float*)d_in[12];
    const float* W_hr   = (const float*)d_in[13];
    const float* b_hr   = (const float*)d_in[14];
    const float* ln2_g  = (const float*)d_in[15];
    const float* ln2_b  = (const float*)d_in[16];
    const float* W_si   = (const float*)d_in[17];
    const float* b_si   = (const float*)d_in[18];
    float* out = (float*)d_out;

    const int N = in_sizes[0] / IN_F;   // 50000 (divisible by 16)
    const int E = in_sizes[1];          // 800000

    // -------- workspace carve-out (256B aligned slices) --------
    char* w = (char*)d_ws;
    auto carve = [&](size_t bytes) -> void* {
        void* p = (void*)w;
        w += (bytes + 255) & ~(size_t)255;
        return p;
    };
    float*    hn    = (float*)carve((size_t)N * IN_F * 4);
    float*    ft    = (float*)carve((size_t)N * OUT_F * 4);
    float*    pOut  = (float*)carve((size_t)N * IN_F * 4);
    float*    gs    = (float*)carve((size_t)N * GATE_F * 4);
    float*    agg   = (float*)carve((size_t)N * OUT_F * 4);
    int*      pmbits= (int*)carve((size_t)N * IN_F * 4);
    unsigned* mkey  = (unsigned*)carve((size_t)N * HEADS * 4);
    float*    zsum  = (float*)carve((size_t)N * HEADS * 4);
    float*    esc   = (float*)carve((size_t)E * HEADS * 4);
    __bf16*   hnb   = (__bf16*)carve((size_t)N * IN_F * 2);
    __bf16*   WaP   = (__bf16*)carve((size_t)IN_F * OUT_F * 2);
    __bf16*   WpP   = (__bf16*)carve((size_t)IN_F * IN_F * 2);
    __bf16*   WgsP  = (__bf16*)carve((size_t)IN_F * GATE_F * 2);
    __bf16*   WgnP  = (__bf16*)carve((size_t)IN_F * GATE_F * 2);
    __bf16*   WhrP  = (__bf16*)carve((size_t)OUT_F * OUT_F * 2);
    __bf16*   WsiP  = (__bf16*)carve((size_t)OUT_F * OUT_F * 2);

    // -------- zero-init accumulators (capture-legal memsets) --------
    hipMemsetAsync(agg,    0, (size_t)N * OUT_F * 4, stream);
    hipMemsetAsync(pmbits, 0, (size_t)N * IN_F * 4, stream);   // 0 bits == 0.0f (p>=0)
    hipMemsetAsync(mkey,   0, (size_t)N * HEADS * 4, stream);  // key 0 == very negative
    hipMemsetAsync(zsum,   0, (size_t)N * HEADS * 4, stream);

    // -------- weights -> bf16, B-fragment packed order --------
    auto pack = [&](const float* s, __bf16* d, int K, int ncols) {
        int n = K * ncols;
        pack_b_kernel<<<(n + 255) / 256, 256, 0, stream>>>(s, d, K, ncols);
    };
    pack(W_attn,   WaP,  IN_F, OUT_F);
    pack(W_pool,   WpP,  IN_F, IN_F);
    pack(W_gself,  WgsP, IN_F, GATE_F);
    pack(W_gneigh, WgnP, IN_F, GATE_F);
    pack(W_hr,     WhrP, OUT_F, OUT_F);
    pack(W_si,     WsiP, OUT_F, OUT_F);

    // -------- pipeline --------
    ln1_kernel<<<(N + 7) / 8, 256, 0, stream>>>(h, ln1_g, ln1_b, hn, hnb, N);

    const int rowBlocks = N / 16;
    gemm_front_kernel<<<rowBlocks, 256, 0, stream>>>(hnb, WaP, WpP, WgsP, b_pool,
                                                     ft, pOut, gs);

    edge_pass1_kernel<<<(E + 7) / 8, 256, 0, stream>>>(src, dst, ft, pOut, esc, mkey,
                                                       pmbits, E);

    long EH = (long)E * HEADS;
    edge_pass2_kernel<<<(int)((EH + 255) / 256), 256, 0, stream>>>(dst, esc, mkey, zsum, EH);

    edge_pass3_kernel<<<(E + 7) / 8, 256, 0, stream>>>(src, dst, ft, esc, zsum, agg, E);

    node_epilogue_kernel<<<rowBlocks, 256, 0, stream>>>(
        hn, agg, pmbits, gs, WgnP, b_gate, W_gr, b_gr, WhrP, b_hr, ln2_g, ln2_b,
        WsiP, b_si, out);
}
